// RegionProposalNetwork_23450521436768
// MI455X (gfx1250) — matrix operand order repaired and checked by reading, not compiled, not run on hardware
//
#include <hip/hip_runtime.h>
#include <math.h>

typedef __bf16 bf16_t;
typedef __attribute__((ext_vector_type(16))) __bf16 v16bf;
typedef __attribute__((ext_vector_type(8)))  __bf16 v8bf;
typedef __attribute__((ext_vector_type(8)))  float  v8f;

// ---------------- problem constants ----------------
#define BATCH 16
#define CIN   512
#define HH    32
#define WW    32
#define AANCH 9
#define NBOX  (HH*WW*AANCH)      // 9216
#define MTOT  (BATCH*HH*WW)      // 16384
#define KT1   144                // (9*512)/32 K-tiles for conv3x3 GEMM
#define NT1   32                 // 512/16 N-tiles
#define KT2   16                 // 512/32 K-tiles for head GEMM
#define NPAD  48                 // 9+36 padded to 3 N-tiles

// ---------------- d_out layout (floats, reference tuple order) ----------------
#define LEN_LOGITS (BATCH*AANCH*HH*WW)           // 147456
#define LEN_BBOX   (BATCH*4*AANCH*HH*WW)         // 589824
#define OFF_LOGITS 0
#define OFF_BBOX   (OFF_LOGITS + LEN_LOGITS)
#define OFF_PROP   (OFF_BBOX   + LEN_BBOX)       // proposals [B,N,4]
#define OFF_KEEP   (OFF_PROP   + BATCH*NBOX*4)   // keep [B,N]
#define OFF_ANCH   (OFF_KEEP   + BATCH*NBOX)     // anchors_b [B,N,4]

// ---------------- workspace sizes (bytes) ----------------
#define XPAD_BYTES ((size_t)BATCH*34*34*CIN*2)   // padded NHWC bf16 input
#define HB_BYTES   ((size_t)MTOT*CIN*2)          // hidden NHWC bf16
#define BP1_BYTES  ((size_t)KT1*NT1*32*16*2)     // packed conv1 weights
#define BP2_BYTES  ((size_t)KT2*3*32*16*2)       // packed head weights
#define SC_BYTES   ((size_t)BATCH*NBOX*4)        // sigmoid scores (A,H,W flat)

// ============================================================================
// pack_x: NCHW f32 -> padded NHWC bf16, LDS tile transpose (coalesced both ways)
// grid: (16 ci-tiles, 32 rows, 16 batch), block 256
// ============================================================================
__global__ __launch_bounds__(256) void rpn_pack_x(const float* __restrict__ x,
                                                  bf16_t* __restrict__ Xpad) {
  __shared__ float tile[32][33];
  const int b   = blockIdx.z;
  const int y   = blockIdx.y;
  const int ci0 = blockIdx.x * 32;
  const int t   = threadIdx.x;
  const int px  = t & 31;     // pixel (x) for load phase
  const int cr  = t >> 5;     // 0..7
  const float* src = x + ((size_t)(b * CIN + ci0) * HH + y) * WW;
#pragma unroll
  for (int r = 0; r < 4; ++r) {
    const int ci = cr + r * 8;
    tile[px][ci] = src[(size_t)ci * (HH * WW) + px];   // consecutive lanes -> consecutive x
  }
  __syncthreads();
  const int cil = t & 31;     // channel for store phase (contiguous in NHWC)
  const int pr  = t >> 5;
  bf16_t* dst = Xpad + (((size_t)(b * 34 + (y + 1)) * 34) + 1) * CIN + ci0 + cil;
#pragma unroll
  for (int r = 0; r < 4; ++r) {
    const int p = pr + r * 8;
    dst[(size_t)p * CIN] = (bf16_t)tile[p][cil];
  }
}

// ============================================================================
// pack_w1: conv1_w [512][512][3][3] f32 -> B-fragment-ordered bf16
// element order: Bp1[(((nt*144 + kt)*32 + lane)*16 + e]; K = kt*32 + (lane>>4)*8 + (e&7) + (e>>3)*16
// ============================================================================
__global__ __launch_bounds__(256) void rpn_pack_w1(const float* __restrict__ w1,
                                                   bf16_t* __restrict__ Bp1) {
  const int t = blockIdx.x * 256 + threadIdx.x;
  const int TOT = NT1 * KT1 * 32 * 16;
  if (t >= TOT) return;
  const int e    = t & 15;
  const int lane = (t >> 4) & 31;
  const int kt   = (t >> 9) % KT1;
  const int nt   = (t >> 9) / KT1;
  const int n    = nt * 16 + (lane & 15);
  const int k    = kt * 32 + (lane >> 4) * 8 + (e & 7) + ((e >> 3) << 4);  // (tap, ci) packed: k = tap*512+ci
  const int tap  = k >> 9;
  const int ci   = k & 511;
  const int ky   = tap / 3, kx = tap % 3;
  Bp1[t] = (bf16_t)w1[(((size_t)n * CIN + ci) * 3 + ky) * 3 + kx];
}

// ============================================================================
// pack_w2: cls_w [9][512] + bbox_w [36][512] -> padded N=48 B-fragments
// ============================================================================
__global__ __launch_bounds__(256) void rpn_pack_w2(const float* __restrict__ clsw,
                                                   const float* __restrict__ bboxw,
                                                   bf16_t* __restrict__ Bp2) {
  const int t = blockIdx.x * 256 + threadIdx.x;
  const int TOT = 3 * KT2 * 32 * 16;
  if (t >= TOT) return;
  const int e    = t & 15;
  const int lane = (t >> 4) & 31;
  const int kt   = (t >> 9) % KT2;
  const int nt   = (t >> 9) / KT2;
  const int n    = nt * 16 + (lane & 15);
  const int k    = kt * 32 + (lane >> 4) * 8 + (e & 7) + ((e >> 3) << 4);  // ci
  float w = 0.f;
  if (n < AANCH)            w = clsw[(size_t)n * CIN + k];
  else if (n < AANCH * 5)   w = bboxw[(size_t)(n - AANCH) * CIN + k];
  Bp2[t] = (bf16_t)w;
}

// ============================================================================
// gemm1: implicit-GEMM 3x3 conv, M=16384, N=512, K=4608, bf16 WMMA, f32 acc.
// block = 256 (8 waves); wave (mt, nt) -> 16x16 tile. Bias+ReLU fused, store bf16 NHWC.
// ============================================================================
__global__ __launch_bounds__(256) void rpn_gemm1(const bf16_t* __restrict__ Xpad,
                                                 const bf16_t* __restrict__ Bp1,
                                                 const float*  __restrict__ bias,
                                                 bf16_t* __restrict__ Hb) {
  const int wave = threadIdx.x >> 5;
  const int lane = threadIdx.x & 31;
  const int mt = blockIdx.x;                  // 0..1023
  const int nt = blockIdx.y * 8 + wave;       // 0..31
  const int m0 = mt * 16;
  const int n0 = nt * 16;
  const int ln   = lane & 15;
  const int half = lane >> 4;
  const int m = m0 + ln;                      // A row for this lane
  const int b = m >> 10;
  const int y = (m >> 5) & 31;
  const int x = m & 31;

  v8f acc = {};
  const bf16_t* bp = Bp1 + ((size_t)(nt * KT1) * 32 + lane) * 16;
#pragma unroll 1
  for (int tap = 0; tap < 9; ++tap) {
    const int ky = tap / 3, kx = tap % 3;
    const bf16_t* arow =
        Xpad + (size_t)((b * 34 + (y + ky)) * 34 + (x + kx)) * CIN;
#pragma unroll 4
    for (int kc = 0; kc < 16; ++kc) {
      const int ci = kc * 32 + half * 8;      // A layout: K runs [h*8,h*8+8) & [16+h*8,...)
      union { v16bf v; v8bf h[2]; } a, bb;
      a.h[0] = *reinterpret_cast<const v8bf*>(arow + ci);
      a.h[1] = *reinterpret_cast<const v8bf*>(arow + ci + 16);
      const bf16_t* bptr = bp + (size_t)(tap * 16 + kc) * 32 * 16;
      bb.h[0] = *reinterpret_cast<const v8bf*>(bptr);
      bb.h[1] = *reinterpret_cast<const v8bf*>(bptr + 8);
      acc = __builtin_amdgcn_wmma_f32_16x16x32_bf16(false, a.v, false, bb.v,
                                                    (short)0, acc, false, false);
    }
  }
  const float bv = bias[n0 + ln];
#pragma unroll
  for (int v = 0; v < 8; ++v) {               // D: m = v + 8*half, n = lane&15
    float r = acc[v] + bv;
    r = r > 0.f ? r : 0.f;
    Hb[(size_t)(m0 + v + 8 * half) * CIN + n0 + ln] = (bf16_t)r;
  }
}

// ============================================================================
// gemm2: both heads fused. M=16384, N=48 (9 cls + 36 bbox + pad), K=512.
// Writes logits/bbox NCHW f32 directly into d_out (bias fused).
// ============================================================================
__global__ __launch_bounds__(32) void rpn_gemm2(const bf16_t* __restrict__ Hb,
                                                const bf16_t* __restrict__ Bp2,
                                                const float* __restrict__ clsb,
                                                const float* __restrict__ bboxb,
                                                float* __restrict__ out) {
  const int lane = threadIdx.x & 31;
  const int mt = blockIdx.x;
  const int nt = blockIdx.y;                  // 0..2
  const int m0 = mt * 16;
  const int ln   = lane & 15;
  const int half = lane >> 4;
  const bf16_t* arow = Hb + (size_t)(m0 + ln) * CIN;
  const bf16_t* bp = Bp2 + ((size_t)(nt * KT2) * 32 + lane) * 16;

  v8f acc = {};
#pragma unroll 4
  for (int kt = 0; kt < KT2; ++kt) {
    const int ci = kt * 32 + half * 8;
    union { v16bf v; v8bf h[2]; } a, bb;
    a.h[0] = *reinterpret_cast<const v8bf*>(arow + ci);
    a.h[1] = *reinterpret_cast<const v8bf*>(arow + ci + 16);
    const bf16_t* bptr = bp + (size_t)kt * 32 * 16;
    bb.h[0] = *reinterpret_cast<const v8bf*>(bptr);
    bb.h[1] = *reinterpret_cast<const v8bf*>(bptr + 8);
    acc = __builtin_amdgcn_wmma_f32_16x16x32_bf16(false, a.v, false, bb.v,
                                                  (short)0, acc, false, false);
  }
  const int n = nt * 16 + ln;
#pragma unroll
  for (int v = 0; v < 8; ++v) {
    const int m = m0 + v + 8 * half;
    const int b = m >> 10, y = (m >> 5) & 31, x = m & 31;
    const float r = acc[v];
    if (n < AANCH) {
      out[OFF_LOGITS + (size_t)(b * AANCH + n) * (HH * WW) + y * WW + x] = r + clsb[n];
    } else if (n < AANCH * 5) {
      out[OFF_BBOX + (size_t)(b * 4 * AANCH + (n - AANCH)) * (HH * WW) + y * WW + x] =
          r + bboxb[n - AANCH];
    }
  }
}

// ============================================================================
// post: anchors (replicating np.round = round-half-even via rintf), decode
// (including the reference's double cxcywh->xyxy quirk), sigmoid scores in
// the reference's (A,H,W) flat order. Writes proposals + anchors_b sections.
// ============================================================================
__global__ __launch_bounds__(256) void rpn_post(float* __restrict__ out,
                                                float* __restrict__ Sc) {
  const int g = blockIdx.x * 256 + threadIdx.x;
  if (g >= BATCH * NBOX) return;
  const int b = g / NBOX;
  const int i = g % NBOX;

  // scores: prob = sigmoid(logits).reshape(B,-1) -> same flat index as logits
  const float lg = out[OFF_LOGITS + g];
  Sc[g] = 1.f / (1.f + expf(-lg));

  // anchor for proposal index i = (y*W + x)*A + a
  const int a = i % AANCH;
  const int pix = i / AANCH;
  const int y = pix >> 5, x = pix & 31;
  const int si = a / 3, ri = a % 3;
  const float s = (float)(128 << si);
  const float r = (ri == 0) ? 0.5f : (ri == 1 ? 1.f : 2.f);
  const float wv = rintf(sqrtf(s * s / r));   // np.round: half-to-even
  const float hv = rintf(wv * r);
  const float xc = (wv - 1.f) * 0.5f, yc = (hv - 1.f) * 0.5f;
  const float sx = (float)x * 16.f + 8.f, sy = (float)y * 16.f + 8.f;
  const float a0 = sx - xc, a1 = sy - yc, a2 = sx + xc, a3 = sy + yc;

  float* anch = out + OFF_ANCH + (size_t)g * 4;
  anch[0] = a0; anch[1] = a1; anch[2] = a2; anch[3] = a3;

  // deltas: bbox.transpose(0,2,3,1) -> channel = a*4 + c at (y,x)
  const size_t bb = OFF_BBOX + (size_t)b * (4 * AANCH * HH * WW) + (size_t)(y * WW + x);
  const float d0 = out[bb + (size_t)(a * 4 + 0) * (HH * WW)];
  const float d1 = out[bb + (size_t)(a * 4 + 1) * (HH * WW)];
  const float d2 = out[bb + (size_t)(a * 4 + 2) * (HH * WW)];
  const float d3 = out[bb + (size_t)(a * 4 + 3) * (HH * WW)];

  const float wa = a2 - a0, ha = a3 - a1;
  const float cx = a0 + 0.5f * wa, cy = a1 + 0.5f * ha;
  const float pcx = d0 * wa + cx, pcy = d1 * ha + cy;
  const float pw = expf(d2) * wa, ph = expf(d3) * ha;
  const float p0 = pcx - 0.5f * pw, p1 = pcy - 0.5f * ph;
  const float p2 = pcx + 0.5f * pw, p3 = pcy + 0.5f * ph;
  float* pr = out + OFF_PROP + (size_t)g * 4;
  pr[0] = p0 - 0.5f * p2; pr[1] = p1 - 0.5f * p3;
  pr[2] = p0 + 0.5f * p2; pr[3] = p1 + 0.5f * p3;
}

// ============================================================================
// nms: one block per batch. Bitonic sort (score desc, invalid=-inf) in LDS,
// then greedy suppression (serial outer over sorted order, parallel inner),
// exactly matching the reference's scan semantics. Scatter keep back.
// Dynamic LDS: 16384*4 (key) + 16384*4 (idx) + 16384 (keep) = 147456 B.
// ============================================================================
__global__ __launch_bounds__(1024) void rpn_nms(const float* __restrict__ Sc,
                                                float* __restrict__ out) {
  extern __shared__ char smem[];
  float* key = (float*)smem;                         // [16384]
  int* sidx  = (int*)(smem + 16384 * 4);             // [16384]
  unsigned char* keep = (unsigned char*)(smem + 16384 * 8);  // [16384]
  const int b = blockIdx.x;
  const int tid = threadIdx.x;
  const int N = NBOX, NP = 16384;
  const float NEG = -__builtin_inff();

  for (int t = tid; t < NP; t += 1024) {
    if (t < N) {
      const float sc = Sc[(size_t)b * N + t];
      key[t] = (sc > 0.5f) ? sc : NEG;               // valid = prob > SCORE_THR
      sidx[t] = t;
    } else { key[t] = NEG; sidx[t] = -1; }
  }
  __syncthreads();

  // bitonic sort, descending
  for (int k = 2; k <= NP; k <<= 1) {
    for (int j = k >> 1; j > 0; j >>= 1) {
      for (int t = tid; t < NP; t += 1024) {
        const int ixj = t ^ j;
        if (ixj > t) {
          const bool up = (t & k) == 0;
          const float av = key[t], cv = key[ixj];
          const bool sw = up ? (av < cv) : (av > cv);
          if (sw) {
            key[t] = cv; key[ixj] = av;
            const int tmp = sidx[t]; sidx[t] = sidx[ixj]; sidx[ixj] = tmp;
          }
        }
      }
      __syncthreads();
    }
  }

  for (int t = tid; t < NP; t += 1024) keep[t] = (key[t] != NEG) ? 1 : 0;
  __syncthreads();

  const float* prop = out + OFF_PROP + (size_t)b * N * 4;
  for (int i = 0; i < N; ++i) {
    if (key[i] == NEG) break;                        // sorted: rest invalid (uniform)
    if (keep[i]) {                                   // uniform (shared)
      const int oi = sidx[i];
      const float x1 = prop[oi * 4 + 0], y1 = prop[oi * 4 + 1];
      const float x2 = prop[oi * 4 + 2], y2 = prop[oi * 4 + 3];
      const float ai = (x2 - x1) * (y2 - y1);
      for (int jj = i + 1 + tid; jj < N; jj += 1024) {
        if (!keep[jj]) continue;
        const int oj = sidx[jj];
        const float bx1 = prop[oj * 4 + 0], by1 = prop[oj * 4 + 1];
        const float bx2 = prop[oj * 4 + 2], by2 = prop[oj * 4 + 3];
        const float xx1 = fmaxf(x1, bx1), yy1 = fmaxf(y1, by1);
        const float xx2 = fminf(x2, bx2), yy2 = fminf(y2, by2);
        const float inter = fmaxf(xx2 - xx1, 0.f) * fmaxf(yy2 - yy1, 0.f);
        const float aj = (bx2 - bx1) * (by2 - by1);
        if (inter / (ai + aj - inter) > 0.7f) keep[jj] = 0;
      }
    }
    __syncthreads();
  }

  for (int t = tid; t < NP; t += 1024) {
    const int o = sidx[t];
    if (o >= 0) out[OFF_KEEP + (size_t)b * N + o] = keep[t] ? 1.0f : 0.0f;
  }
}

// ============================================================================
extern "C" void kernel_launch(void* const* d_in, const int* in_sizes, int n_in,
                              void* d_out, int out_size, void* d_ws, size_t ws_size,
                              hipStream_t stream) {
  (void)in_sizes; (void)n_in; (void)out_size; (void)ws_size;
  const float* x      = (const float*)d_in[0];
  const float* conv1w = (const float*)d_in[1];
  const float* conv1b = (const float*)d_in[2];
  const float* clsw   = (const float*)d_in[3];
  const float* clsb   = (const float*)d_in[4];
  const float* bboxw  = (const float*)d_in[5];
  const float* bboxb  = (const float*)d_in[6];
  float* out = (float*)d_out;

  char* ws = (char*)d_ws;
  size_t off = 0;
  auto carve = [&](size_t bytes) {
    char* p = ws + off;
    off = (off + bytes + 255) & ~(size_t)255;
    return p;
  };
  bf16_t* Xpad = (bf16_t*)carve(XPAD_BYTES);
  bf16_t* Hb   = (bf16_t*)carve(HB_BYTES);
  bf16_t* Bp1  = (bf16_t*)carve(BP1_BYTES);
  bf16_t* Bp2  = (bf16_t*)carve(BP2_BYTES);
  float*  Sc   = (float*) carve(SC_BYTES);

  hipMemsetAsync(Xpad, 0, XPAD_BYTES, stream);  // zero padded border (capturable)

  rpn_pack_x<<<dim3(16, 32, 16), 256, 0, stream>>>(x, Xpad);
  rpn_pack_w1<<<(NT1 * KT1 * 32 * 16 + 255) / 256, 256, 0, stream>>>(conv1w, Bp1);
  rpn_pack_w2<<<(3 * KT2 * 32 * 16 + 255) / 256, 256, 0, stream>>>(clsw, bboxw, Bp2);
  rpn_gemm1<<<dim3(MTOT / 16, NT1 / 8), 256, 0, stream>>>(Xpad, Bp1, conv1b, Hb);
  rpn_gemm2<<<dim3(MTOT / 16, 3), 32, 0, stream>>>(Hb, Bp2, clsb, bboxb, out);
  rpn_post<<<(BATCH * NBOX + 255) / 256, 256, 0, stream>>>(out, Sc);
  rpn_nms<<<BATCH, 1024, 16384 * 9, stream>>>(Sc, out);
}
